// VQModule_22531398435504
// MI455X (gfx1250) — matrix operand order repaired and checked by reading, not compiled, not run on hardware
//
#include <hip/hip_runtime.h>
#include <hip/hip_bf16.h>

// ---------------------------------------------------------------------------
// VQ quantize: out[b,c,h,w] = embed[argmin_k ||x[b,:,h,w]-e_k||^2, c]
//              ids[b*4096 + w*64 + h] = argmin (note transposed h/w)
// B=32, D=64, H=W=64, K=512, N=131072 pixels.
// Score s = f.e - 0.5*||e||^2, argmax over k.  bf16 hi/lo split GEMM via
// v_wmma_f32_16x16x32_bf16 (3 WMMAs per 32-deep chunk, D=64 -> 6 per tile).
// ---------------------------------------------------------------------------

typedef __attribute__((ext_vector_type(16))) __bf16       v16bf;
typedef __attribute__((ext_vector_type(8)))  float        v8f;
typedef __attribute__((ext_vector_type(4)))  unsigned int v4u;

union Frag { v4u q[2]; v16bf v; };

__device__ inline unsigned short f2bf(float x) {           // RNE f32 -> bf16
    unsigned int u = __float_as_uint(x);
    unsigned int r = u + 0x7FFFu + ((u >> 16) & 1u);
    return (unsigned short)(r >> 16);
}
__device__ inline float bf2f(unsigned short h) {
    return __uint_as_float(((unsigned int)h) << 16);
}

__device__ inline v8f wmma_bf16(v16bf a, v16bf b, v8f c) {
    return __builtin_amdgcn_wmma_f32_16x16x32_bf16(
        /*neg_a=*/false, a, /*neg_b=*/false, b,
        /*c_mod=*/(short)0, c, /*reuse_a=*/false, /*reuse_b=*/false);
}

// -------------------------- prep: split codebook ---------------------------
// embed[512][64] fp32 -> ehi/elo bf16 (row-major) + cbias[k] = -0.5*||e_k||^2
__global__ __launch_bounds__(64) void vq_prep(const float* __restrict__ embed,
                                              unsigned short* __restrict__ ehi,
                                              unsigned short* __restrict__ elo,
                                              float* __restrict__ cbias) {
    __shared__ float red[64];
    const int k = blockIdx.x, c = threadIdx.x;
    float e = embed[k * 64 + c];
    unsigned short hi = f2bf(e);
    float lo = e - bf2f(hi);
    ehi[k * 64 + c] = hi;
    elo[k * 64 + c] = f2bf(lo);
    red[c] = e * e;
    __syncthreads();
    for (int s = 32; s > 0; s >>= 1) {
        if (c < s) red[c] += red[c + s];
        __syncthreads();
    }
    if (c == 0) cbias[k] = -0.5f * red[0];
}

// ------------------------------- main kernel -------------------------------
// grid = 1024 blocks (32 tiles/image * 32 images), 256 threads = 8 waves.
// Tile = 128 consecutive (h,w) pixels of one image; wave w owns 16 pixels.
__global__ __launch_bounds__(256) void vq_main(const float* __restrict__ x,
                                               const float* __restrict__ embed,
                                               const unsigned short* __restrict__ ehi,
                                               const unsigned short* __restrict__ elo,
                                               const float* __restrict__ cbias,
                                               float* __restrict__ out,
                                               float* __restrict__ idsOut) {
    // padded stride 72 (144 B, 16B-aligned rows) breaks LDS bank conflicts
    __shared__ unsigned short es_hi[256 * 72];   // 36.0 KB  codebook chunk hi
    __shared__ unsigned short es_lo[256 * 72];   // 36.0 KB  codebook chunk lo
    __shared__ unsigned short xs_hi[128 * 72];   // 18.0 KB  pixel tile hi
    __shared__ unsigned short xs_lo[128 * 72];   // 18.0 KB  pixel tile lo
    __shared__ float cb_s[512];
    __shared__ int   ids_s[128];

    const int t = threadIdx.x;
    const int tile = blockIdx.x;            // 0..1023
    const int b = tile >> 5;                // image
    const int hwBase = (tile & 31) * 128;   // flat h*64+w base
    const size_t xBase = (size_t)b * (64 * 4096) + hwBase;

    // ---- stage x tile: coalesced fp32 reads, bf16 hi/lo into LDS ----
    for (int it = 0; it < 32; ++it) {
        int f = it * 256 + t;               // f = c*128 + p
        int c = f >> 7, p = f & 127;
        float v = x[xBase + (size_t)c * 4096 + p];
        unsigned short hi = f2bf(v);
        xs_hi[p * 72 + c] = hi;
        xs_lo[p * 72 + c] = f2bf(v - bf2f(hi));
    }
    for (int i = t; i < 512; i += 256) cb_s[i] = cbias[i];

    const int wave = t >> 5;
    const int lane = t & 31;
    const int m0   = wave * 16;             // wave's first pixel in tile
    const int half = lane >> 4;
    const int mr   = lane & 15;

    __syncthreads();

    // ---- A fragments (16x32 bf16 layout: lanes<16 K=0..7,16..23; +16: 8..15,24..31)
    Frag a0h, a0l, a1h, a1l;
    {
        const int arow = (m0 + mr) * 72;
        const int c0 = half * 8;
        a0h.q[0] = *(const v4u*)&xs_hi[arow + c0];
        a0h.q[1] = *(const v4u*)&xs_hi[arow + c0 + 16];
        a1h.q[0] = *(const v4u*)&xs_hi[arow + c0 + 32];
        a1h.q[1] = *(const v4u*)&xs_hi[arow + c0 + 48];
        a0l.q[0] = *(const v4u*)&xs_lo[arow + c0];
        a0l.q[1] = *(const v4u*)&xs_lo[arow + c0 + 16];
        a1l.q[0] = *(const v4u*)&xs_lo[arow + c0 + 32];
        a1l.q[1] = *(const v4u*)&xs_lo[arow + c0 + 48];
    }

    float best[8];
    int   bid[8];
#pragma unroll
    for (int r = 0; r < 8; ++r) { best[r] = -3.4e38f; bid[r] = 0; }

    // ---- two passes of 256 codes each ----
    for (int kp = 0; kp < 2; ++kp) {
        const int kOff = kp * 256;
        // stage codebook chunk (dword copies, coalesced)
        for (int it = t; it < 256 * 32; it += 256) {
            int row = it >> 5, d = it & 31;
            *(unsigned int*)&es_hi[row * 72 + d * 2] =
                ((const unsigned int*)ehi)[(kOff + row) * 32 + d];
            *(unsigned int*)&es_lo[row * 72 + d * 2] =
                ((const unsigned int*)elo)[(kOff + row) * 32 + d];
        }
        __syncthreads();

        for (int kt = 0; kt < 16; ++kt) {
            const int k0 = kt * 16;
            const int nG = kOff + k0 + mr;  // this lane's code column
            // B fragments (32x16 bf16: lanes<16 K=0..7,16..23; lanes>=16 K=8..15,24..31; col=lane&15)
            const int brow = (k0 + mr) * 72;
            const int cB = half * 8;
            Frag b0h, b1h, b0l, b1l;
            b0h.q[0] = *(const v4u*)&es_hi[brow + cB];
            b0h.q[1] = *(const v4u*)&es_hi[brow + cB + 16];
            b1h.q[0] = *(const v4u*)&es_hi[brow + 32 + cB];
            b1h.q[1] = *(const v4u*)&es_hi[brow + 32 + cB + 16];
            b0l.q[0] = *(const v4u*)&es_lo[brow + cB];
            b0l.q[1] = *(const v4u*)&es_lo[brow + cB + 16];
            b1l.q[0] = *(const v4u*)&es_lo[brow + 32 + cB];
            b1l.q[1] = *(const v4u*)&es_lo[brow + 32 + cB + 16];

            v8f acc = {};
            acc = wmma_bf16(a0h.v, b0h.v, acc);   // hi*hi (d 0..31)
            acc = wmma_bf16(a1h.v, b1h.v, acc);   // hi*hi (d 32..63)
            acc = wmma_bf16(a0h.v, b0l.v, acc);   // hi*lo
            acc = wmma_bf16(a1h.v, b1l.v, acc);
            acc = wmma_bf16(a0l.v, b0h.v, acc);   // lo*hi
            acc = wmma_bf16(a1l.v, b1h.v, acc);

            const float cbv = cb_s[nG];
#pragma unroll
            for (int r = 0; r < 8; ++r) {         // acc[r] -> pixel m0+8*half+r
                float s = acc[r] + cbv;
                if (s > best[r]) { best[r] = s; bid[r] = nG; }
            }
        }
        __syncthreads();   // before restaging es / after everyone done reading
    }

    // ---- argmax reduce across the 16 lanes of each half (ties -> lower id)
#pragma unroll
    for (int m = 1; m < 16; m <<= 1) {
#pragma unroll
        for (int r = 0; r < 8; ++r) {
            float os = __shfl_xor(best[r], m, 32);
            int   oi = __shfl_xor(bid[r],  m, 32);
            if (os > best[r] || (os == best[r] && oi < bid[r])) {
                best[r] = os; bid[r] = oi;
            }
        }
    }
    if (mr == 0) {
#pragma unroll
        for (int r = 0; r < 8; ++r) ids_s[m0 + half * 8 + r] = bid[r];
    }
    __syncthreads();

    // ---- write out (coalesced stores; embed gather stays L2-resident) ----
    for (int it = 0; it < 32; ++it) {
        int f = it * 256 + t;
        int c = f >> 7, p = f & 127;
        out[xBase + (size_t)c * 4096 + p] = embed[ids_s[p] * 64 + c];
    }
    // ids output is (h,w)-transposed: flat = b*4096 + w*64 + h
    if (t < 128) {
        int g = hwBase + t;
        int h = g >> 6, w = g & 63;
        idsOut[(size_t)b * 4096 + w * 64 + h] = (float)ids_s[t];
    }
}

// ------------------------------- launcher ----------------------------------
extern "C" void kernel_launch(void* const* d_in, const int* in_sizes, int n_in,
                              void* d_out, int out_size, void* d_ws, size_t ws_size,
                              hipStream_t stream) {
    (void)in_sizes; (void)n_in; (void)out_size; (void)ws_size;
    const float* x     = (const float*)d_in[0];   // [32,64,64,64] fp32
    const float* embed = (const float*)d_in[1];   // [512,64] fp32

    unsigned short* ehi   = (unsigned short*)d_ws;            // 64 KB
    unsigned short* elo   = ehi + 512 * 64;                   // 64 KB
    float*          cbias = (float*)(elo + 512 * 64);         // 2 KB

    float* out    = (float*)d_out;                            // 8388608 floats
    float* idsOut = out + (size_t)32 * 64 * 64 * 64;          // 131072 entries

    vq_prep<<<512, 64, 0, stream>>>(embed, ehi, elo, cbias);
    vq_main<<<1024, 256, 0, stream>>>(x, embed, ehi, elo, cbias, out, idsOut);
}
